// TopkRouter_25074019074698
// MI455X (gfx1250) — matrix operand order repaired and checked by reading, not compiled, not run on hardware
//
#include <hip/hip_runtime.h>
#include <hip/hip_bf16.h>
#include <math.h>

// ---------------------------------------------------------------------------
// Top-K MoE router, fused: score = relu(x@w1 + b1) @ w2 + b2 ; top2 ; softmax
//   x : [32768, 1024] f32   w1 : [1024, 4096] f32   w2 : [4096, 8] f32
// GEMM1 runs on v_wmma_f32_16x16x32_bf16 (CDNA5); h never hits HBM.
// 256 threads / 8 waves per WG -> 4 waves per SIMD32 at 2 WGs/WGP (LDS-bound).
// ---------------------------------------------------------------------------

#define EMBED   1024
#define HIDDEN  4096
#define NEXP    8
#define ROWS    64           // tokens per workgroup
#define WAVES   8            // 256 threads
#define NC_PER_WAVE 32       // 256 H-chunks / 8 waves
#define A_STRIDE 1032        // ushorts per LDS A row (2064 B; 516 dw % 64 = 4 banks skew)

typedef __attribute__((ext_vector_type(16))) __bf16 v16bf;
typedef __attribute__((ext_vector_type(8)))  float  v8f;
typedef __attribute__((ext_vector_type(4)))  unsigned int u32x4;

union Frag {
    v16bf h;
    u32x4 q[2];
};

__device__ __forceinline__ unsigned short f2bf(float f) {
    unsigned int u = __float_as_uint(f);
    return (unsigned short)((u + 0x7FFFu + ((u >> 16) & 1u)) >> 16);
}
__device__ __forceinline__ unsigned int pack2bf(float a, float b) {
    return (unsigned int)f2bf(a) | ((unsigned int)f2bf(b) << 16);
}

// ---------------------------------------------------------------------------
// Kernel 1: convert w1 f32 -> bf16 in WMMA B-fragment order.
// Fragment f = nc*32 + kc is a 32x16 (KxN) tile, 512 bf16, laid out [lane][16]:
//   lane 0..15  : N = lane,    K = kc*32 + 0..15   (contiguous)
//   lane 16..31 : N = lane-16, K = kc*32 + 16..31  (contiguous)
// ---------------------------------------------------------------------------
__global__ void convert_w1_kernel(const float* __restrict__ w1,
                                  unsigned short* __restrict__ ws) {
    int gid = blockIdx.x * blockDim.x + threadIdx.x;
    if (gid >= EMBED * HIDDEN) return;
    int frag = gid >> 9;          // which 512-elem fragment
    int w    = gid & 511;
    int lane = w >> 4;
    int i    = w & 15;
    int nc   = frag >> 5;
    int kc   = frag & 31;
    int K = kc * 32 + ((lane >> 4) << 4) + i;
    int N = (nc << 4) + (lane & 15);
    ws[gid] = f2bf(w1[(size_t)K * HIDDEN + N]);
}

// ---------------------------------------------------------------------------
// Kernel 2: fused router.
// ---------------------------------------------------------------------------
__global__ void __launch_bounds__(WAVES * 32)
router_kernel(const float* __restrict__ inputs,
              const float* __restrict__ b1,
              const float* __restrict__ w2,
              const float* __restrict__ b2,
              const unsigned short* __restrict__ wsW1,
              float* __restrict__ out,
              int tokens) {
    __shared__ unsigned short As[ROWS * A_STRIDE];   // 132096 B, bf16 A tile
    __shared__ float part[WAVES][ROWS][NEXP];        // 16384 B, score partials

    const int tid   = threadIdx.x;
    const int lane  = tid & 31;
    const int wv    = tid >> 5;
    const int lhalf = lane >> 4;       // 0 or 1
    const int l15   = lane & 15;
    const size_t rowbase = (size_t)blockIdx.x * ROWS;

    // ---- Stage A tile: 64 rows x 1024 f32 -> bf16 in LDS (padded rows) ----
    const float* inp = inputs + rowbase * EMBED;
    for (int j = tid * 4; j < ROWS * EMBED; j += WAVES * 32 * 4) {
        int row = j >> 10;
        int col = j & (EMBED - 1);
        const float4 v = *(const float4*)(inp + row * EMBED + col);
        uint2 p;
        p.x = pack2bf(v.x, v.y);
        p.y = pack2bf(v.z, v.w);
        *(uint2*)(&As[row * A_STRIDE + col]) = p;
    }
    __syncthreads();

    // ---- Main loop: wave wv covers H-chunks [wv*32, wv*32+32) for all rows --
    const unsigned short* arow = &As[(l15)*A_STRIDE + (lhalf << 4)];
    for (int mt = 0; mt < 4; ++mt) {
        const int m0 = mt << 4;
        const unsigned short* arowm = arow + m0 * A_STRIDE;
        float sacc[8][NEXP];
        #pragma unroll
        for (int r = 0; r < 8; ++r)
            #pragma unroll
            for (int e = 0; e < NEXP; ++e) sacc[r][e] = 0.0f;

        for (int nci = 0; nci < NC_PER_WAVE; ++nci) {
            const int nc   = wv * NC_PER_WAVE + nci;
            const int hcol = (nc << 4) + l15;
            const unsigned short* bp = wsW1 + (size_t)nc * 16384 + lane * 16;

            v8f c = {};
            #pragma unroll 4
            for (int kc = 0; kc < 32; ++kc) {
                Frag a, b;
                const unsigned short* ap = arowm + kc * 32;
                a.q[0] = *(const u32x4*)(ap);        // K slab low  (8 bf16)
                a.q[1] = *(const u32x4*)(ap + 8);    // K slab high (8 bf16)
                const unsigned short* bpp = bp + (kc << 9);
                b.q[0] = *(const u32x4*)(bpp);
                b.q[1] = *(const u32x4*)(bpp + 8);
                c = __builtin_amdgcn_wmma_f32_16x16x32_bf16(
                        false, a.h, false, b.h, (short)0, c, false, false);
            }

            // fold h chunk into score partials: relu(h + b1) @ w2
            const float  bb  = b1[hcol];
            const float4 wlo = *(const float4*)(w2 + (size_t)hcol * NEXP);
            const float4 whi = *(const float4*)(w2 + (size_t)hcol * NEXP + 4);
            #pragma unroll
            for (int r = 0; r < 8; ++r) {
                float hv = c[r] + bb;
                hv = hv > 0.0f ? hv : 0.0f;
                sacc[r][0] += hv * wlo.x;  sacc[r][1] += hv * wlo.y;
                sacc[r][2] += hv * wlo.z;  sacc[r][3] += hv * wlo.w;
                sacc[r][4] += hv * whi.x;  sacc[r][5] += hv * whi.y;
                sacc[r][6] += hv * whi.z;  sacc[r][7] += hv * whi.w;
            }
        }

        // reduce across the 16-lane N-group (lanes 0-15: rows m0+r; 16-31: m0+8+r)
        #pragma unroll
        for (int r = 0; r < 8; ++r) {
            #pragma unroll
            for (int e = 0; e < NEXP; ++e) {
                float v = sacc[r][e];
                v += __shfl_xor(v, 1, 32);
                v += __shfl_xor(v, 2, 32);
                v += __shfl_xor(v, 4, 32);
                v += __shfl_xor(v, 8, 32);
                sacc[r][e] = v;
            }
        }
        if (l15 == 0) {
            #pragma unroll
            for (int r = 0; r < 8; ++r)
                #pragma unroll
                for (int e = 0; e < NEXP; ++e)
                    part[wv][m0 + (lhalf << 3) + r][e] = sacc[r][e];
        }
    }
    __syncthreads();

    // ---- Epilogue: one thread per token: top-2 + masked softmax ----
    if (tid < ROWS) {
        float s[NEXP];
        #pragma unroll
        for (int e = 0; e < NEXP; ++e) {
            float acc = b2[e];
            #pragma unroll
            for (int w = 0; w < WAVES; ++w) acc += part[w][tid][e];
            s[e] = acc;
        }

        int   i1 = 0;    float v1 = s[0];
        #pragma unroll
        for (int e = 1; e < NEXP; ++e) if (s[e] > v1) { v1 = s[e]; i1 = e; }
        int   i2 = -1;   float v2 = -INFINITY;
        #pragma unroll
        for (int e = 0; e < NEXP; ++e)
            if (e != i1 && s[e] > v2) { v2 = s[e]; i2 = e; }

        const float e2  = __expf(v2 - v1);
        const float inv = 1.0f / (1.0f + e2);
        const float p1  = inv;
        const float p2  = e2 * inv;

        const size_t token = rowbase + tid;
        float* rout = out + token * NEXP;
        #pragma unroll
        for (int e = 0; e < NEXP; ++e)
            rout[e] = (e == i1) ? p1 : ((e == i2) ? p2 : 0.0f);

        float* ridx = out + (size_t)tokens * NEXP + token * 2;
        ridx[0] = (float)i1;
        ridx[1] = (float)i2;
    }
}

extern "C" void kernel_launch(void* const* d_in, const int* in_sizes, int n_in,
                              void* d_out, int out_size, void* d_ws, size_t ws_size,
                              hipStream_t stream) {
    (void)n_in; (void)out_size; (void)ws_size;
    const float* inputs = (const float*)d_in[0];
    const float* w1     = (const float*)d_in[1];
    const float* b1     = (const float*)d_in[2];
    const float* w2     = (const float*)d_in[3];
    const float* b2     = (const float*)d_in[4];
    unsigned short* wsW1 = (unsigned short*)d_ws;   // 1024*4096 bf16 = 8 MB
    float* out = (float*)d_out;

    const int tokens = in_sizes[0] / EMBED;         // 32768

    const int nW1 = EMBED * HIDDEN;
    convert_w1_kernel<<<(nW1 + 255) / 256, 256, 0, stream>>>(w1, wsW1);

    router_kernel<<<tokens / ROWS, WAVES * 32, 0, stream>>>(
        inputs, b1, w2, b2, wsW1, out, tokens);
}
